// TransformerEncoder_46952582480353
// MI455X (gfx1250) — matrix operand order repaired and checked by reading, not compiled, not run on hardware
//
#include <hip/hip_runtime.h>
#include <hip/hip_fp16.h>
#include <stdint.h>

typedef __attribute__((ext_vector_type(16))) _Float16 v16h;
typedef __attribute__((ext_vector_type(8)))  _Float16 v8h;
typedef __attribute__((ext_vector_type(8)))  float    v8f;

#define B_   8
#define S_   2048
#define F_   8
#define H_   8
#define D_   64
#define HID_ 512
#define NTOT (B_*S_*F_)          /* 131072 */
#define EPS_ 1e-5f
// softmax scale (1/sqrt(D)) * log2(e), folded into the Q projection
#define QSCALE_ (0.125f * 1.44269504088896340736f)

__device__ inline v16h cat8(v8h lo, v8h hi) {
  return __builtin_shufflevector(lo, hi, 0,1,2,3,4,5,6,7,8,9,10,11,12,13,14,15);
}

// direct v_max_num_f32: avoids the compiler's sNaN-canonicalize self-max
__device__ inline float vmaxf(float a, float b) {
  float d;
  asm("v_max_num_f32 %0, %1, %2" : "=v"(d) : "v"(a), "v"(b));
  return d;
}

// ---- VALU DPP 16-lane-row reductions (row_ror:N = dpp_ctrl 0x120+N) -------
template<int N>
__device__ inline float row_ror(float x) {
  return __int_as_float(__builtin_amdgcn_update_dpp(
      0, __float_as_int(x), 0x120 + N, 0xF, 0xF, true));
}
__device__ inline float rowmax16(float x) {
  x = vmaxf(x, row_ror<1>(x));
  x = vmaxf(x, row_ror<2>(x));
  x = vmaxf(x, row_ror<4>(x));
  x = vmaxf(x, row_ror<8>(x));
  return x;
}
__device__ inline float rowsum16(float x) {
  x += row_ror<1>(x);
  x += row_ror<2>(x);
  x += row_ror<4>(x);
  x += row_ror<8>(x);
  return x;
}

// ---------------------------------------------------------------------------
// Kernel 1: per-head Q/K/V projections (F=8 -> D=64), fp32 -> fp16.
// qh is pre-scaled by (1/sqrt(D))*log2(e) so attention WMMA scores need no
// per-element scaling.  qh, khn stored [bh][s][d]; vh stored transposed
// [bh][d][s] so the P*V WMMA B-fragment is a contiguous LDS read.
// ---------------------------------------------------------------------------
__global__ void __launch_bounds__(256) proj_kernel(
    const float* __restrict__ q, const float* __restrict__ k, const float* __restrict__ v,
    const float* __restrict__ Wq, const float* __restrict__ bq,
    const float* __restrict__ Wk, const float* __restrict__ bk,
    const float* __restrict__ Wv, const float* __restrict__ bv,
    _Float16* __restrict__ qh, _Float16* __restrict__ khn, _Float16* __restrict__ vhT)
{
  int idx = blockIdx.x * 256 + threadIdx.x;          // over B*H*S*D
  if (idx >= B_*H_*S_*D_) return;
  int d =  idx        & (D_-1);
  int s = (idx >> 6)  & (S_-1);
  int h = (idx >> 17) & (H_-1);
  int b =  idx >> 20;

  const float* qrow = q + ((size_t)b*S_ + s)*F_;
  const float* krow = k + ((size_t)b*S_ + s)*F_;
  const float* vrow = v + ((size_t)b*S_ + s)*F_;

  float aq = bq[h*D_ + d], ak = bk[h*D_ + d], av = bv[h*D_ + d];
#pragma unroll
  for (int f = 0; f < F_; ++f) {
    aq += qrow[f] * Wq[(h*F_ + f)*D_ + d];
    ak += krow[f] * Wk[(h*F_ + f)*D_ + d];
    av += vrow[f] * Wv[(h*F_ + f)*D_ + d];
  }
  qh [(size_t)idx] = (_Float16)(aq * QSCALE_);
  khn[(size_t)idx] = (_Float16)ak;
  vhT[((size_t)(b*H_ + h)*D_ + d)*S_ + s] = (_Float16)av;
}

// ---------------------------------------------------------------------------
// Kernel 2: flash attention. Block = 256 threads (8 waves), each wave owns a
// 16-row Q tile of one (b,h); iterate T in steps of 32 with online softmax
// in the log2 domain.  K/V tiles double-buffered in LDS via gfx1250 async
// global->LDS loads so the next tile's transfer hides under the 8 WMMAs.
// ---------------------------------------------------------------------------
__global__ void __launch_bounds__(256) attn_kernel(
    const _Float16* __restrict__ qh, const _Float16* __restrict__ khn,
    const _Float16* __restrict__ vhT, float* __restrict__ concat)
{
  __shared__ _Float16 Kt[2][32][72];     // [buf][t_local][d], padded
  __shared__ _Float16 Vt[2][64][40];     // [buf][d][t_local], padded
  __shared__ _Float16 Pt[8][16][40];     // per-wave P tile [m][k], padded

  const int tid  = threadIdx.x;
  const int wave = tid >> 5;
  const int lane = tid & 31;
  const int hi   = lane >> 4;            // lane half (ISA fragment layouts)
  const int ln   = lane & 15;

  const int stile = blockIdx.x & 15;     // S/128 tiles
  const int h     = (blockIdx.x >> 4) & 7;
  const int b     =  blockIdx.x >> 7;
  const int bh    = b*H_ + h;
  const int s0    = stile*128 + wave*16;

  // --- Q A-fragments (16x32 f16, two chunks over D=64), contiguous k order (same as B):
  // lane L: m = L&15, kbase = (L>>4)*16; a[i]=A[m][kb+i]
  const _Float16* qr = qh + ((size_t)bh*S_ + s0 + ln)*D_ + hi*16;
  v16h Aq0 = cat8(*(const v8h*)(qr +  0), *(const v8h*)(qr +  8));
  v16h Aq1 = cat8(*(const v8h*)(qr + 32), *(const v8h*)(qr + 40));

  v8f acc0 = {}, acc1 = {}, acc2 = {}, acc3 = {};
  float mrun[8], lsum[8];                // lsum: per-lane partial row sums
#pragma unroll
  for (int r = 0; r < 8; ++r) { mrun[r] = -__builtin_inff(); lsum[r] = 0.f; }

  const int kr = tid >> 3, kc = (tid & 7)*8;   // this thread's K-tile slice
  const int vr = tid >> 2, vc = (tid & 3)*8;   // this thread's V-tile slice
  const _Float16* ksrc = khn + ((size_t)bh*S_ + kr)*D_ + kc;
  const _Float16* vsrc = vhT + ((size_t)bh*D_ + vr)*S_ + vc;
  uint32_t kdst[2] = { (uint32_t)(uintptr_t)&Kt[0][kr][kc],
                       (uint32_t)(uintptr_t)&Kt[1][kr][kc] };
  uint32_t vdst[2] = { (uint32_t)(uintptr_t)&Vt[0][vr][vc],
                       (uint32_t)(uintptr_t)&Vt[1][vr][vc] };

  // prologue: start tile 0 into buffer 0
  asm volatile("global_load_async_to_lds_b128 %0, %1, off"
               :: "v"(kdst[0]), "v"(ksrc) : "memory");
  asm volatile("global_load_async_to_lds_b128 %0, %1, off"
               :: "v"(vdst[0]), "v"(vsrc) : "memory");

  const int NIT = S_/32;
  int cur = 0;
  for (int it = 0; it < NIT; ++it) {
    __syncthreads();            // readers of buf cur^1 (prev iter) are done
    if (it + 1 < NIT) {         // start streaming tile it+1 into other buffer
      const _Float16* ks = ksrc + (size_t)(it + 1) * 32 * D_;
      const _Float16* vs = vsrc + (it + 1) * 32;
      asm volatile("global_load_async_to_lds_b128 %0, %1, off"
                   :: "v"(kdst[cur ^ 1]), "v"(ks) : "memory");
      asm volatile("global_load_async_to_lds_b128 %0, %1, off"
                   :: "v"(vdst[cur ^ 1]), "v"(vs) : "memory");
      // async loads complete in order: <=2 outstanding => tile it has landed
      asm volatile("s_wait_asynccnt 0x2" ::: "memory");
    } else {
      asm volatile("s_wait_asynccnt 0x0" ::: "memory");
    }
    __syncthreads();            // tile it visible to all waves

    // ---- scores: two 16x16 tiles (cols t0..t0+15 = X, t0+16..31 = Y)
    // B-fragment: lane L: n = L&15, kbase = (L>>4)*16; b[i] = K[t0+n][kb+i]
    const _Float16* kx = &Kt[cur][ln     ][hi*16];
    const _Float16* ky = &Kt[cur][16 + ln][hi*16];
    v16h Bx0 = cat8(*(const v8h*)(kx +  0), *(const v8h*)(kx +  8));
    v16h Bx1 = cat8(*(const v8h*)(kx + 32), *(const v8h*)(kx + 40));
    v16h By0 = cat8(*(const v8h*)(ky +  0), *(const v8h*)(ky +  8));
    v16h By1 = cat8(*(const v8h*)(ky + 32), *(const v8h*)(ky + 40));

    v8f zz = {};
    v8f sx = __builtin_amdgcn_wmma_f32_16x16x32_f16(false, Aq0, false, Bx0, (short)0, zz, false, false);
    sx     = __builtin_amdgcn_wmma_f32_16x16x32_f16(false, Aq1, false, Bx1, (short)0, sx, false, false);
    v8f sy = __builtin_amdgcn_wmma_f32_16x16x32_f16(false, Aq0, false, By0, (short)0, zz, false, false);
    sy     = __builtin_amdgcn_wmma_f32_16x16x32_f16(false, Aq1, false, By1, (short)0, sy, false, false);

    // ---- online softmax (scores already scaled into log2 domain via Q)
#pragma unroll
    for (int r = 0; r < 8; ++r) {
      float x0 = sx[r], y0 = sy[r];
      float t  = rowmax16(vmaxf(x0, y0));          // VALU DPP, no DS traffic
      float mn = vmaxf(mrun[r], t);
      float corr = exp2f(mrun[r] - mn);
      float px   = exp2f(x0 - mn);
      float py   = exp2f(y0 - mn);
      mrun[r] = mn;
      lsum[r] = lsum[r]*corr + (px + py);          // per-lane partial, no reduce
      acc0[r] *= corr; acc1[r] *= corr; acc2[r] *= corr; acc3[r] *= corr;
      // C-layout -> LDS (row-major 16x32 P tile, fp16)
      const int row = hi*8 + r;
      Pt[wave][row][ln]      = (_Float16)px;
      Pt[wave][row][16 + ln] = (_Float16)py;
    }
    asm volatile("s_wait_dscnt 0x0" ::: "memory");   // intra-wave LDS RAW

    // ---- P A-fragment (16x32) from LDS, then ctx += P x V per d-chunk
    const _Float16* pr = &Pt[wave][ln][hi*16];
    v16h Ap = cat8(*(const v8h*)(pr + 0), *(const v8h*)(pr + 8));

#pragma unroll
    for (int c = 0; c < 4; ++c) {
      const _Float16* vv = &Vt[cur][c*16 + ln][hi*16];
      v16h Bv = cat8(*(const v8h*)(vv + 0), *(const v8h*)(vv + 8));
      v8f* acc = (c == 0) ? &acc0 : (c == 1) ? &acc1 : (c == 2) ? &acc2 : &acc3;
      *acc = __builtin_amdgcn_wmma_f32_16x16x32_f16(false, Ap, false, Bv, (short)0, *acc, false, false);
    }
    cur ^= 1;
  }

  // ---- finalize: reduce per-lane row sums once, normalize, write ctx
#pragma unroll
  for (int r = 0; r < 8; ++r) {
    float l = rowsum16(lsum[r]);                    // single end-of-loop reduce
    float inv = 1.0f / l;
    int s = s0 + hi*8 + r;
    float* crow = concat + ((size_t)b*S_ + s)*(H_*D_) + h*D_;
    crow[ 0 + ln] = acc0[r]*inv;
    crow[16 + ln] = acc1[r]*inv;
    crow[32 + ln] = acc2[r]*inv;
    crow[48 + ln] = acc3[r]*inv;
  }
}

// ---------------------------------------------------------------------------
// Kernel 3: mha = concat @ Wo + bo ; x = q + mha.  One block per (b,s) row.
// ---------------------------------------------------------------------------
__global__ void __launch_bounds__(256) wo_kernel(
    const float* __restrict__ concat, const float* __restrict__ Wo,
    const float* __restrict__ bo, const float* __restrict__ q, float* __restrict__ x)
{
  __shared__ float red[256][8];
  const int row = blockIdx.x;
  const int tid = threadIdx.x;
  const float* crow = concat + (size_t)row * (H_*D_);
  float p[8] = {0,0,0,0,0,0,0,0};
  for (int c = tid; c < H_*D_; c += 256) {
    float cv = crow[c];
#pragma unroll
    for (int f = 0; f < 8; ++f) p[f] += cv * Wo[c*8 + f];
  }
#pragma unroll
  for (int f = 0; f < 8; ++f) red[tid][f] = p[f];
  __syncthreads();
  for (int st = 128; st > 0; st >>= 1) {
    if (tid < st)
#pragma unroll
      for (int f = 0; f < 8; ++f) red[tid][f] += red[tid + st][f];
    __syncthreads();
  }
  if (tid < 8)
    x[(size_t)row*8 + tid] = q[(size_t)row*8 + tid] + red[0][tid] + bo[tid];
}

// ---------------------------------------------------------------------------
// Kernel 4: global (full-tensor) mean / inv_std for the LayerNorms.
// ---------------------------------------------------------------------------
__global__ void __launch_bounds__(256) reduce_kernel(
    const float* __restrict__ x, float* __restrict__ stats, int which)
{
  __shared__ float ssum[256], ssq[256];
  const int tid = threadIdx.x;
  float s = 0.f, q2 = 0.f;
  for (int i = tid; i < NTOT; i += 256) { float v = x[i]; s += v; q2 += v*v; }
  ssum[tid] = s; ssq[tid] = q2;
  __syncthreads();
  for (int st = 128; st > 0; st >>= 1) {
    if (tid < st) { ssum[tid] += ssum[tid+st]; ssq[tid] += ssq[tid+st]; }
    __syncthreads();
  }
  if (tid == 0) {
    float mean = ssum[0] / (float)NTOT;
    float var  = ssq[0] / (float)NTOT - mean*mean;
    stats[which*2]     = mean;
    stats[which*2 + 1] = rsqrtf(var + EPS_);
  }
}

// ---------------------------------------------------------------------------
// Kernel 5: LN1 -> FFN (8 -> 512 -> 8) -> z = q + ffn.  Block per (b,s).
// ---------------------------------------------------------------------------
__global__ void __launch_bounds__(256) ffn_kernel(
    const float* __restrict__ x, const float* __restrict__ stats,
    const float* __restrict__ g1, const float* __restrict__ bln1,
    const float* __restrict__ W1, const float* __restrict__ b1,
    const float* __restrict__ W2, const float* __restrict__ b2,
    const float* __restrict__ q, float* __restrict__ z)
{
  __shared__ float hn[8];
  __shared__ float red[256][8];
  const int row = blockIdx.x;
  const int tid = threadIdx.x;
  const float mean = stats[0], istd = stats[1];
  if (tid < 8) {
    size_t i = (size_t)row*8 + tid;
    hn[tid] = (x[i] - mean) * istd * g1[i] + bln1[i];
  }
  __syncthreads();
  float hr[8];
#pragma unroll
  for (int f = 0; f < 8; ++f) hr[f] = hn[f];

  float p[8] = {0,0,0,0,0,0,0,0};
  for (int u = tid; u < HID_; u += 256) {
    float a = b1[u];
#pragma unroll
    for (int f = 0; f < 8; ++f) a += hr[f] * W1[f*HID_ + u];
    a = fmaxf(a, 0.f);                      // ReLU
#pragma unroll
    for (int f = 0; f < 8; ++f) p[f] += a * W2[u*8 + f];
  }
#pragma unroll
  for (int f = 0; f < 8; ++f) red[tid][f] = p[f];
  __syncthreads();
  for (int st = 128; st > 0; st >>= 1) {
    if (tid < st)
#pragma unroll
      for (int f = 0; f < 8; ++f) red[tid][f] += red[tid + st][f];
    __syncthreads();
  }
  if (tid < 8) {
    size_t i = (size_t)row*8 + tid;
    z[i] = q[i] + red[0][tid] + b2[tid];
  }
}

// ---------------------------------------------------------------------------
// Kernel 6: final LN2, replicate output 3x (reference returns (out,out,out)).
// ---------------------------------------------------------------------------
__global__ void __launch_bounds__(256) final_kernel(
    const float* __restrict__ zz, const float* __restrict__ stats,
    const float* __restrict__ g2, const float* __restrict__ bln2,
    float* __restrict__ out)
{
  int i = blockIdx.x*256 + threadIdx.x;
  if (i >= NTOT) return;
  float mean = stats[2], istd = stats[3];
  float o = (zz[i] - mean) * istd * g2[i] + bln2[i];
  out[i] = o; out[i + NTOT] = o; out[i + 2*NTOT] = o;
}

// ---------------------------------------------------------------------------
extern "C" void kernel_launch(void* const* d_in, const int* in_sizes, int n_in,
                              void* d_out, int out_size, void* d_ws, size_t ws_size,
                              hipStream_t stream)
{
  const float* q    = (const float*)d_in[0];
  const float* k    = (const float*)d_in[1];
  const float* v    = (const float*)d_in[2];
  const float* Wq   = (const float*)d_in[3];
  const float* bq   = (const float*)d_in[4];
  const float* Wk   = (const float*)d_in[5];
  const float* bk   = (const float*)d_in[6];
  const float* Wv   = (const float*)d_in[7];
  const float* bv   = (const float*)d_in[8];
  const float* Wo   = (const float*)d_in[9];
  const float* bo   = (const float*)d_in[10];
  const float* g1   = (const float*)d_in[11];
  const float* bl1  = (const float*)d_in[12];
  const float* W1   = (const float*)d_in[13];
  const float* b1   = (const float*)d_in[14];
  const float* W2   = (const float*)d_in[15];
  const float* b2   = (const float*)d_in[16];
  const float* g2   = (const float*)d_in[17];
  const float* bl2  = (const float*)d_in[18];
  float* out = (float*)d_out;

  // workspace layout (bytes, all 256-aligned)
  char* ws = (char*)d_ws;
  const size_t nBHSD = (size_t)B_*H_*S_*D_;            // 8,388,608
  _Float16* qh     = (_Float16*)(ws);                          // 16 MB
  _Float16* khn    = (_Float16*)(ws + nBHSD*2);                // 16 MB
  _Float16* vhT    = (_Float16*)(ws + nBHSD*4);                // 16 MB
  float*    concat = (float*)   (ws + nBHSD*6);                // 32 MB
  float*    x      = (float*)   (ws + nBHSD*6 + nBHSD*4);      // 512 KB
  float*    z      = (float*)   (ws + nBHSD*6 + nBHSD*4 + (size_t)NTOT*4);
  float*    stats  = (float*)   (ws + nBHSD*6 + nBHSD*4 + (size_t)NTOT*8);

  proj_kernel  <<<(int)(nBHSD/256), 256, 0, stream>>>(q, k, v, Wq, bq, Wk, bk, Wv, bv, qh, khn, vhT);
  attn_kernel  <<<B_*H_*(S_/128),   256, 0, stream>>>(qh, khn, vhT, concat);
  wo_kernel    <<<B_*S_,            256, 0, stream>>>(concat, Wo, bo, q, x);
  reduce_kernel<<<1,                256, 0, stream>>>(x, stats, 0);
  ffn_kernel   <<<B_*S_,            256, 0, stream>>>(x, stats, g1, bl1, W1, b1, W2, b2, q, z);
  reduce_kernel<<<1,                256, 0, stream>>>(z, stats, 1);
  final_kernel <<<(NTOT+255)/256,   256, 0, stream>>>(z, stats, g2, bl2, out);
  (void)in_sizes; (void)n_in; (void)out_size; (void)ws_size;
}
